// PNP_35802847380227
// MI455X (gfx1250) — matrix operand order repaired and checked by reading, not compiled, not run on hardware
//
#include <hip/hip_runtime.h>
#include <hip/hip_bf16.h>
#include <math.h>

// ---------------- problem constants ----------------
#define BB 64
#define PP 196
#define DD 768
#define NCLS 200
#define CC 201            // N_CLASSES + 1
#define KP 5
#define NN (BB * PP)      // 12544 patches
#define NCOL (CC * KP)    // 1005 logits columns
#define NCOL_PAD 1024     // padded B rows for the WMMA GEMM
#define GAMMA 0.999f
#define INV_TEMP 5.0f     // 1/0.2
#define INV_EPS_SK 20.0f  // 1/0.05

// GEMM tiling
#define KCH 64            // K elements per LDS chunk
#define NCHUNK (DD / KCH) // 12 chunks (even -> 2x unrolled ping-pong loop)
#define ASTRIDE 72        // padded LDS row stride in bf16 elems (144B -> conflict-free)

typedef __attribute__((ext_vector_type(16))) __bf16 v16bf;
typedef __attribute__((ext_vector_type(8)))  __bf16 bf16x8;
typedef __attribute__((ext_vector_type(8)))  float  v8f;

// ---------------- row L2-normalize: f32 -> bf16 ----------------
__global__ void k_norm_bf16(const float* __restrict__ src, __bf16* __restrict__ dst) {
    __shared__ float red[256];
    const int row = blockIdx.x, tid = threadIdx.x;
    const float* s = src + (size_t)row * DD;
    float x0 = s[tid], x1 = s[tid + 256], x2 = s[tid + 512];
    red[tid] = x0 * x0 + x1 * x1 + x2 * x2;
    __syncthreads();
    for (int st = 128; st > 0; st >>= 1) {
        if (tid < st) red[tid] += red[tid + st];
        __syncthreads();
    }
    const float scale = 1.0f / fmaxf(sqrtf(red[0]), 1e-12f);
    __bf16* d = dst + (size_t)row * DD;
    d[tid]       = (__bf16)(x0 * scale);
    d[tid + 256] = (__bf16)(x1 * scale);
    d[tid + 512] = (__bf16)(x2 * scale);
}

// prototypes: rows [0,1005) normalized, rows [1005,1024) zero padded
__global__ void k_norm_proto_bf16(const float* __restrict__ src, __bf16* __restrict__ dst) {
    __shared__ float red[256];
    const int row = blockIdx.x, tid = threadIdx.x;
    __bf16* d = dst + (size_t)row * DD;
    if (row >= NCOL) {  // uniform per block
        d[tid] = (__bf16)0.0f; d[tid + 256] = (__bf16)0.0f; d[tid + 512] = (__bf16)0.0f;
        return;
    }
    const float* s = src + (size_t)row * DD;
    float x0 = s[tid], x1 = s[tid + 256], x2 = s[tid + 512];
    red[tid] = x0 * x0 + x1 * x1 + x2 * x2;
    __syncthreads();
    for (int st = 128; st > 0; st >>= 1) {
        if (tid < st) red[tid] += red[tid + st];
        __syncthreads();
    }
    const float scale = 1.0f / fmaxf(sqrtf(red[0]), 1e-12f);
    d[tid]       = (__bf16)(x0 * scale);
    d[tid + 256] = (__bf16)(x1 * scale);
    d[tid + 512] = (__bf16)(x2 * scale);
}

// row L2-normalize: f32 -> f32 (raw patch tokens; feeds prototype update GEMM)
__global__ void k_norm_f32(const float* __restrict__ src, float* __restrict__ dst) {
    __shared__ float red[256];
    const int row = blockIdx.x, tid = threadIdx.x;
    const float* s = src + (size_t)row * DD;
    float x0 = s[tid], x1 = s[tid + 256], x2 = s[tid + 512];
    red[tid] = x0 * x0 + x1 * x1 + x2 * x2;
    __syncthreads();
    for (int st = 128; st > 0; st >>= 1) {
        if (tid < st) red[tid] += red[tid + st];
        __syncthreads();
    }
    const float scale = 1.0f / fmaxf(sqrtf(red[0]), 1e-12f);
    float* d = dst + (size_t)row * DD;
    d[tid] = x0 * scale; d[tid + 256] = x1 * scale; d[tid + 512] = x2 * scale;
}

// ---------------- WMMA GEMM helpers ----------------
// one 64-K chunk: 2 K-steps x 4 M-subtiles = 8 WMMAs.
// All 4 A fragments of a K-step are ds_loaded into distinct registers BEFORE the 4 WMMAs,
// so the scheduler can issue partial s_wait_dscnt instead of a full stall per WMMA.
__device__ __forceinline__ void mma_chunk(const __bf16* __restrict__ sbuf,
                                          int l15, int lhalf,
                                          const v16bf* __restrict__ bfrag,  // [2]
                                          v8f* __restrict__ acc) {          // [4]
#pragma unroll
    for (int ks = 0; ks < 2; ++ks) {
        v16bf af[4];
#pragma unroll
        for (int r = 0; r < 4; ++r) {
            const __bf16* ap = sbuf + (r * 16 + l15) * ASTRIDE + ks * 32 + lhalf * 16;
            bf16x8 lo = *(const bf16x8*)(ap);
            bf16x8 hi = *(const bf16x8*)(ap + 8);
            af[r] = __builtin_shufflevector(lo, hi, 0, 1, 2, 3, 4, 5, 6, 7,
                                            8, 9, 10, 11, 12, 13, 14, 15);
        }
#pragma unroll
        for (int r = 0; r < 4; ++r)
            acc[r] = __builtin_amdgcn_wmma_f32_16x16x32_bf16(
                false, af[r], false, bfrag[ks], (short)0, acc[r], false, false);
    }
}

__device__ __forceinline__ void load_bfrag(const __bf16* __restrict__ bCol, int chunk,
                                           v16bf* __restrict__ bfrag) {
    bfrag[0] = *(const v16bf*)(bCol + chunk * KCH);
    bfrag[1] = *(const v16bf*)(bCol + chunk * KCH + 32);
}

// ---------------- main WMMA GEMM: logits[N, 1005] = pt_bf16 @ proto_bf16^T ----------------
// Block = 256 threads (8 waves). Block tile 64(M) x 128(N); wave tile 64(M) x 16(N).
// A tile staged through double-buffered LDS (one fetch per block instead of per wave);
// both the cooperative A staging loads AND the per-wave B fragments are pipelined one
// chunk ahead (2x-unrolled ping-pong), so each chunk's 8 WMMAs start without waiting on
// loads issued in the same chunk. LDS rows padded to 144B -> conflict-free ds_load_b128.
__global__ void k_gemm_wmma(const __bf16* __restrict__ A,   // [NN, DD]
                            const __bf16* __restrict__ Bm,  // [NCOL_PAD, DD]
                            float* __restrict__ C) {        // [NN, NCOL]
    __shared__ __bf16 sA[2][64 * ASTRIDE];

    const int tid   = threadIdx.x;
    const int lane  = tid & 31;
    const int wave  = tid >> 5;
    const int l15   = lane & 15;
    const int lhalf = lane >> 4;                 // 0 or 1
    const int mbase = blockIdx.x * 64;
    const int j0    = blockIdx.y * 128 + wave * 16;

    // cooperative staging: 64 rows x 64 K = 512 x 16B; 2 pieces per thread
    const int srow = tid >> 3;        // 0..31  (second piece: +32)
    const int sgrp = tid & 7;         // 8-elem group within the 64-K chunk
    const __bf16* gA0 = A + (size_t)(mbase + srow)      * DD + sgrp * 8;
    const __bf16* gA1 = A + (size_t)(mbase + srow + 32) * DD + sgrp * 8;
    const int soff0 = srow * ASTRIDE + sgrp * 8;
    const int soff1 = soff0 + 32 * ASTRIDE;

    const __bf16* bCol = Bm + (size_t)(j0 + l15) * DD + lhalf * 16;  // 16 contiguous K

    v8f acc[4] = {v8f{}, v8f{}, v8f{}, v8f{}};

    // prologue: stage A chunk 0, preload B fragments for chunk 0
    {
        bf16x8 t0 = *(const bf16x8*)(gA0);
        bf16x8 t1 = *(const bf16x8*)(gA1);
        *(bf16x8*)(&sA[0][soff0]) = t0;
        *(bf16x8*)(&sA[0][soff1]) = t1;
    }
    v16bf bEven[2], bOdd[2];
    load_bfrag(bCol, 0, bEven);
    __syncthreads();

    for (int i = 0; i < NCHUNK; i += 2) {
        // ---- even chunk i: compute from sA[0] with bEven ----
        bf16x8 n0 = *(const bf16x8*)(gA0 + (i + 1) * KCH);   // stage chunk i+1
        bf16x8 n1 = *(const bf16x8*)(gA1 + (i + 1) * KCH);
        load_bfrag(bCol, i + 1, bOdd);                       // B frags chunk i+1

        mma_chunk(&sA[0][0], l15, lhalf, bEven, acc);

        *(bf16x8*)(&sA[1][soff0]) = n0;
        *(bf16x8*)(&sA[1][soff1]) = n1;
        __syncthreads();

        // ---- odd chunk i+1: compute from sA[1] with bOdd ----
        const bool more = (i + 2) < NCHUNK;   // block-uniform
        if (more) {
            n0 = *(const bf16x8*)(gA0 + (i + 2) * KCH);      // stage chunk i+2
            n1 = *(const bf16x8*)(gA1 + (i + 2) * KCH);
            load_bfrag(bCol, i + 2, bEven);                  // B frags chunk i+2
        }

        mma_chunk(&sA[1][0], l15, lhalf, bOdd, acc);

        if (more) {
            *(bf16x8*)(&sA[0][soff0]) = n0;
            *(bf16x8*)(&sA[0][soff1]) = n1;
        }
        __syncthreads();
    }

    const int col = j0 + l15;
    if (col < NCOL) {
#pragma unroll
        for (int r = 0; r < 4; ++r) {
            const int mrow = mbase + r * 16 + lhalf * 8;
#pragma unroll
            for (int v = 0; v < 8; ++v)
                C[(size_t)(mrow + v) * NCOL + col] = acc[r][v];
        }
    }
}

// ---------------- image_logits[b, col] = max over 196 patches ----------------
__global__ void k_img_max(const float* __restrict__ logits, float* __restrict__ img) {
    const int col = blockIdx.x * 256 + threadIdx.x;
    const int b = blockIdx.y;
    if (col >= NCOL) return;
    const float* p = logits + (size_t)b * PP * NCOL + col;
    float m = -INFINITY;
    for (int i = 0; i < PP; ++i) m = fmaxf(m, p[(size_t)i * NCOL]);
    img[(size_t)b * NCOL + col] = m;
}

// ---------------- class logits: softmax(sa)*K weighted sum / TEMP ----------------
__global__ void k_cls(const float* __restrict__ img, const float* __restrict__ sa,
                      float* __restrict__ out) {
    const int t = blockIdx.x * 256 + threadIdx.x;
    if (t >= BB * NCLS) return;
    const int b = t / NCLS, c = t % NCLS;
    float w[KP], mx = -INFINITY, s = 0.0f;
#pragma unroll
    for (int k = 0; k < KP; ++k) { w[k] = sa[c * KP + k]; mx = fmaxf(mx, w[k]); }
#pragma unroll
    for (int k = 0; k < KP; ++k) { w[k] = expf(w[k] - mx); s += w[k]; }
    const float wk = (float)KP / s;
    float acc = 0.0f;
#pragma unroll
    for (int k = 0; k < KP; ++k)
        acc += img[(size_t)b * NCOL + c * KP + k] * (w[k] * wk);
    out[t] = acc * INV_TEMP;
}

// ---------------- Sinkhorn (compact Q[N,5]) ----------------
__global__ void k_zero_i32(int* __restrict__ p, int n) {
    const int t = blockIdx.x * 256 + threadIdx.x;
    if (t < n) p[t] = 0;
}

// Q init + per-class counts (per-class global /sum cancels inside the first row-norm)
__global__ void k_sk_init(const float* __restrict__ logits, const int* __restrict__ lab,
                          float* __restrict__ Q, int* __restrict__ cnt) {
    const int t = blockIdx.x * 256 + threadIdx.x;
    if (t >= NN) return;
    const int c = lab[t];
    const float* l = logits + (size_t)t * NCOL + c * KP;
#pragma unroll
    for (int k = 0; k < KP; ++k) Q[t * KP + k] = expf(l[k] * INV_EPS_SK);
    atomicAdd(&cnt[c], 1);
}

__global__ void k_scan(const int* __restrict__ cnt, int* __restrict__ offs) {
    if (threadIdx.x == 0 && blockIdx.x == 0) {
        int s = 0;
        for (int c = 0; c < CC; ++c) { offs[c] = s; s += cnt[c]; }
        offs[CC] = s;
    }
}

__global__ void k_scatter(const int* __restrict__ lab, const int* __restrict__ offs,
                          int* __restrict__ cursor, int* __restrict__ idxlist) {
    const int t = blockIdx.x * 256 + threadIdx.x;
    if (t >= NN) return;
    const int c = lab[t];
    const int pos = atomicAdd(&cursor[c], 1);
    idxlist[offs[c] + pos] = t;
}

__global__ void k_sk_rowacc(const float* __restrict__ Q, const int* __restrict__ lab,
                            float* __restrict__ rowsum) {
    const int t = blockIdx.x * 256 + threadIdx.x;
    if (t >= NN) return;
    const int c = lab[t];
#pragma unroll
    for (int k = 0; k < KP; ++k) atomicAdd(&rowsum[c * KP + k], Q[t * KP + k]);
}

__global__ void k_sk_update(float* __restrict__ Q, const int* __restrict__ lab,
                            const float* __restrict__ rowsum, const int* __restrict__ cnt,
                            int last) {
    const int t = blockIdx.x * 256 + threadIdx.x;
    if (t >= NN) return;
    const int c = lab[t];
    float q[KP], cs = 0.0f;
#pragma unroll
    for (int k = 0; k < KP; ++k) {
        const float r = rowsum[c * KP + k];
        float v = Q[t * KP + k] / (r > 0.0f ? r : 1.0f) * (1.0f / KP);
        q[k] = v; cs += v;
    }
    float inv = 1.0f / (cs > 0.0f ? cs : 1.0f);
    if (!last) inv /= (float)cnt[c];  // last iter: final *Nc cancels /Nc
#pragma unroll
    for (int k = 0; k < KP; ++k) Q[t * KP + k] = q[k] * inv;
}

__global__ void k_assign(const float* __restrict__ Q, const int* __restrict__ lab,
                         float* __restrict__ out) {
    const int t = blockIdx.x * 256 + threadIdx.x;
    if (t >= NN) return;
    const int c = lab[t];
    float best = -INFINITY; int bi = 0;
#pragma unroll
    for (int k = 0; k < KP; ++k) {
        const float v = Q[t * KP + k];
        if (v > best) { best = v; bi = k; }
    }
    out[t] = (float)(bi + c * KP);  // small int32 values, exact as floats
}

// ---------------- prototype EMA: P_new = gamma*P + (1-gamma)*(Q_c @ I_c) ----------------
__global__ void k_pnew(const float* __restrict__ Q, const float* __restrict__ rpt,
                       const float* __restrict__ proto, const int* __restrict__ idxlist,
                       const int* __restrict__ offs, float* __restrict__ out) {
    const int c = blockIdx.x, tid = threadIdx.x;
    const int beg = offs[c], end = offs[c + 1];
    float acc[KP][3];
#pragma unroll
    for (int k = 0; k < KP; ++k)
#pragma unroll
        for (int j = 0; j < 3; ++j) acc[k][j] = 0.0f;

    for (int i = beg; i < end; ++i) {
        const int n = idxlist[i];
        float q[KP];
#pragma unroll
        for (int k = 0; k < KP; ++k) q[k] = Q[n * KP + k];
        const float* x = rpt + (size_t)n * DD;
#pragma unroll
        for (int j = 0; j < 3; ++j) {
            const float xv = x[tid + j * 256];
#pragma unroll
            for (int k = 0; k < KP; ++k) acc[k][j] += q[k] * xv;
        }
    }
    const bool present = end > beg;
#pragma unroll
    for (int k = 0; k < KP; ++k)
#pragma unroll
        for (int j = 0; j < 3; ++j) {
            const size_t idx = (size_t)(c * KP + k) * DD + tid + j * 256;
            const float p0 = proto[idx];
            out[idx] = present ? GAMMA * p0 + (1.0f - GAMMA) * acc[k][j] : p0;
        }
}

// ---------------- launch ----------------
extern "C" void kernel_launch(void* const* d_in, const int* in_sizes, int n_in,
                              void* d_out, int out_size, void* d_ws, size_t ws_size,
                              hipStream_t stream) {
    const float* patch_tokens = (const float*)d_in[0];
    const float* raw_tokens   = (const float*)d_in[1];
    const float* prototypes   = (const float*)d_in[2];
    const float* sa_weights   = (const float*)d_in[3];
    const int*   labels       = (const int*)d_in[4];

    // outputs (concatenated, all f32)
    float* o_logits = (float*)d_out;                       // [NN, 1005]
    float* o_img    = o_logits + (size_t)NN * NCOL;        // [64, 1005]
    float* o_cls    = o_img + (size_t)BB * NCOL;           // [64, 200]
    float* o_asg    = o_cls + (size_t)BB * NCLS;           // [64, 196]
    float* o_pnew   = o_asg + NN;                          // [201, 5, 768]

    // workspace carve (all chunks 256B-multiples; total ~59.7 MB)
    char* w = (char*)d_ws;
    __bf16* ptb    = (__bf16*)w;            w += (size_t)NN * DD * 2;        // 19.27 MB
    __bf16* protob = (__bf16*)w;            w += (size_t)NCOL_PAD * DD * 2;  // 1.57 MB
    float*  rpt    = (float*)w;             w += (size_t)NN * DD * 4;        // 38.54 MB
    float*  Q      = (float*)w;             w += (size_t)NN * KP * 4;        // 0.25 MB
    float*  rowsum = (float*)w;             w += 1024 * 4;
    int*    cnt    = (int*)w;               w += 256 * 4;
    int*    cursor = (int*)w;               w += 256 * 4;
    int*    offs   = (int*)w;               w += 256 * 4;
    int*    idxlist= (int*)w;               w += (size_t)NN * 4;

    // 1) normalizations
    k_norm_bf16      <<<NN, 256, 0, stream>>>(patch_tokens, ptb);
    k_norm_proto_bf16<<<NCOL_PAD, 256, 0, stream>>>(prototypes, protob);
    k_norm_f32       <<<NN, 256, 0, stream>>>(raw_tokens, rpt);

    // 2) main WMMA GEMM -> logits
    k_gemm_wmma<<<dim3(NN / 64, NCOL_PAD / 128), 256, 0, stream>>>(ptb, protob, o_logits);

    // 3) image max + classifier
    k_img_max<<<dim3(4, BB), 256, 0, stream>>>(o_logits, o_img);
    k_cls<<<(BB * NCLS + 255) / 256, 256, 0, stream>>>(o_img, sa_weights, o_cls);

    // 4) Sinkhorn (compact) + CSR build
    k_zero_i32<<<4, 256, 0, stream>>>(cnt, 768);  // zeros cnt+cursor+offs (contiguous)
    k_sk_init<<<NN / 256, 256, 0, stream>>>(o_logits, labels, Q, cnt);
    k_scan<<<1, 32, 0, stream>>>(cnt, offs);
    k_scatter<<<NN / 256, 256, 0, stream>>>(labels, offs, cursor, idxlist);
    for (int it = 0; it < 3; ++it) {
        k_zero_i32<<<4, 256, 0, stream>>>((int*)rowsum, 1024);
        k_sk_rowacc<<<NN / 256, 256, 0, stream>>>(Q, labels, rowsum);
        k_sk_update<<<NN / 256, 256, 0, stream>>>(Q, labels, rowsum, cnt, it == 2);
    }

    // 5) part assignment + prototype EMA update
    k_assign<<<NN / 256, 256, 0, stream>>>(Q, labels, o_asg);
    k_pnew<<<CC, 256, 0, stream>>>(Q, rpt, prototypes, idxlist, offs, o_pnew);
}
